// ObjectSelector_78692390797622
// MI455X (gfx1250) — matrix operand order repaired and checked by reading, not compiled, not run on hardware
//
#include <hip/hip_runtime.h>
#include <hip/hip_bf16.h>
#include <math.h>

// ---------------------------------------------------------------------------
// Problem constants (match reference)
// ---------------------------------------------------------------------------
#define B_SEG 4096
#define T_OBJ 262144
#define F_DIM 128
#define H_DIM 256
#define C_DIM 512

typedef __attribute__((ext_vector_type(16))) __bf16 v16bf;
typedef __attribute__((ext_vector_type(8)))  float  v8f;
typedef __attribute__((ext_vector_type(4)))  unsigned int u32x4;
typedef __attribute__((ext_vector_type(8)))  int i32x8;
typedef __attribute__((ext_vector_type(4)))  int i32x4;

// ---------------------------------------------------------------------------
// Fragment loaders (wave32 WMMA layouts per CDNA5 ISA 7.12.2)
// ---------------------------------------------------------------------------
// A (16x32 bf16) from LDS tile stored row-major with `strideHalves` pitch.
//   lane<16 : row M=lane,     slots 0-7 -> K=kt*32+0..7,  slots 8-15 -> K=kt*32+16..23
//   lane>=16: row M=lane-16,  slots 0-7 -> K=kt*32+8..15, slots 8-15 -> K=kt*32+24..31
__device__ __forceinline__ v16bf load_A_frag(const __bf16* base, int strideHalves,
                                             int lane, int kt) {
  int row = lane & 15;
  int k0  = kt * 32 + ((lane >> 4) << 3);   // +0 or +8
  const __bf16* p = base + (size_t)row * strideHalves + k0;
  union { float4 f[2]; v16bf v; } u;
  u.f[0] = *(const float4*)(p);        // K = k0 .. k0+7
  u.f[1] = *(const float4*)(p + 16);   // K = k0+16 .. k0+23
  return u.v;
}

// B (32x16 bf16) from pre-swizzled weights: tile (kt,nt) stores 32 lanes x 16
// halves contiguously -> each lane reads its 32 bytes with two float4 loads.
__device__ __forceinline__ v16bf load_B_frag(const __bf16* Wsw, int ktiles,
                                             int kt, int nt, int lane) {
  const __bf16* p = Wsw + (((size_t)(nt * ktiles + kt) * 32 + lane) * 16);
  union { float4 f[2]; v16bf v; } u;
  u.f[0] = *(const float4*)(p);
  u.f[1] = *(const float4*)(p + 8);
  return u.v;
}

__device__ __forceinline__ v8f wmma_bf16(v16bf a, v16bf b, v8f c) {
  return __builtin_amdgcn_wmma_f32_16x16x32_bf16(false, a, false, b,
                                                 (short)0, c, false, false);
}

// ---------------------------------------------------------------------------
// Tensor Data Mover: load a 16 x 128 bf16 tile (row-major) from global into
// LDS with 8-half row padding (pitch 136 halves = 68 dwords).
//   D# group0: count=1, lds_addr, global_addr, type=2
//   D# group1: data_size=2B, pad_enable, pad_interval=5 (64 dw), pad_amount=3
//              (4 dw), tensor_dim0=128, tensor_dim1=T, tile 128x16, stride 128
// ---------------------------------------------------------------------------
__device__ __forceinline__ void tdm_load_x_tile(const __bf16* gsrc, unsigned lds_off) {
  unsigned long long ga = (unsigned long long)(size_t)gsrc;
  unsigned galo = (unsigned)__builtin_amdgcn_readfirstlane((int)(unsigned)(ga & 0xffffffffu));
  unsigned gahi = (unsigned)__builtin_amdgcn_readfirstlane((int)(unsigned)((ga >> 32) & 0x1ffffffu));
  unsigned lo   = (unsigned)__builtin_amdgcn_readfirstlane((int)lds_off);
  u32x4 g0 = { 1u,                      // count=1, is_restore=0, gather off
               lo,                      // lds_addr (bytes)
               galo,                    // global_addr[31:0]
               gahi | (2u << 30) };     // global_addr[56:32] | type=2
  i32x8 g1 = { (int)((1u << 16) | (1u << 20) | (5u << 22) | (3u << 25)),
               (int)(128u << 16),       // tensor_dim0[15:0] = 128
               0,                       // tensor_dim0 hi = 0, tensor_dim1 lo = 0
               (int)((128u << 16) | 4u),// tensor_dim1 hi = 4 (262144), tile_dim0=128
               16,                      // tile_dim1 = 16 rows
               128,                     // tensor_dim0_stride lo = 128
               0, 0 };
  i32x4 z4 = { 0, 0, 0, 0 };
#if defined(__clang_major__) && __clang_major__ >= 23
  i32x8 z8 = { 0, 0, 0, 0, 0, 0, 0, 0 };
  __builtin_amdgcn_tensor_load_to_lds(g0, g1, z4, z4, z8, 0);
#else
  __builtin_amdgcn_tensor_load_to_lds(g0, g1, z4, z4, 0);
#endif
}

// ---------------------------------------------------------------------------
// Kernel A: fp32 -> bf16 row-major convert (vectorized x4)
// ---------------------------------------------------------------------------
__global__ void cvt_bf16_kernel(const float4* __restrict__ src,
                                __bf16* __restrict__ dst, int n4) {
  int i = blockIdx.x * blockDim.x + threadIdx.x;
  if (i >= n4) return;
  float4 f = src[i];
  __bf16 o[4] = { (__bf16)f.x, (__bf16)f.y, (__bf16)f.z, (__bf16)f.w };
  *(uint2*)(dst + (size_t)i * 4) = *(const uint2*)o;
}

// ---------------------------------------------------------------------------
// Kernel 0: fp32 weight [K][N] -> bf16 swizzled B-fragment layout.
//   lane<16 : n=lane,     slots 0-7 -> K = kt*32 + 0..7,  slots 8-15 -> K = kt*32 + 16..23
//   lane>=16: n=lane-16,  slots 0-7 -> K = kt*32 + 8..15, slots 8-15 -> K = kt*32 + 24..31
// ---------------------------------------------------------------------------
__global__ void swizzle_weights_kernel(const float* __restrict__ W,
                                       __bf16* __restrict__ out,
                                       int K, int N) {
  int tid = blockIdx.x * blockDim.x + threadIdx.x;
  int ktiles = K >> 5, ntiles = N >> 4;
  int total = ktiles * ntiles * 32;
  if (tid >= total) return;
  int lane = tid & 31;
  int tile = tid >> 5;
  int kt = tile % ktiles;
  int nt = tile / ktiles;
  int n  = lane & 15;
  int kb = kt * 32 + ((lane >> 4) << 3);
  __bf16* dst = out + (size_t)tid * 16;
#pragma unroll
  for (int i = 0; i < 16; ++i)
    dst[i] = (__bf16)W[(size_t)(kb + (i < 8 ? i : i + 8)) * N + (nt * 16 + n)];
}

// ---------------------------------------------------------------------------
// Kernel 1: q = context @ Wq + bq   [4096 x 512] @ [512 x 256]
// 256 threads (8 waves); block owns 16 rows; wave owns 2 N-tiles; kt-outer
// loop so the A fragment is loaded once per kt and reused for both N tiles.
// ---------------------------------------------------------------------------
__global__ void qgemm_kernel(const float* __restrict__ context,
                             const __bf16* __restrict__ Wqs,
                             const float* __restrict__ bq,
                             float* __restrict__ qout) {
  __shared__ __align__(16) __bf16 Abuf[16][C_DIM + 8];   // 520 halves pitch
  int tid = threadIdx.x, wave = tid >> 5, lane = tid & 31;
  int nlo = lane & 15, mhi = (lane >> 4) << 3;
  int m0 = blockIdx.x * 16;

  for (int i = tid; i < 16 * C_DIM; i += 256) {
    int r = i >> 9, c = i & (C_DIM - 1);
    Abuf[r][c] = (__bf16)context[(size_t)(m0 + r) * C_DIM + c];
  }
  __syncthreads();

  const int ktiles = C_DIM / 32;   // 16
  v8f acc0, acc1;
  {
    float bias0 = bq[(wave * 2 + 0) * 16 + nlo];
    float bias1 = bq[(wave * 2 + 1) * 16 + nlo];
#pragma unroll
    for (int j = 0; j < 8; ++j) { acc0[j] = bias0; acc1[j] = bias1; }
  }
  for (int kt = 0; kt < ktiles; ++kt) {
    v16bf a = load_A_frag(&Abuf[0][0], C_DIM + 8, lane, kt);
    acc0 = wmma_bf16(a, load_B_frag(Wqs, ktiles, kt, wave * 2 + 0, lane), acc0);
    acc1 = wmma_bf16(a, load_B_frag(Wqs, ktiles, kt, wave * 2 + 1, lane), acc1);
  }
#pragma unroll
  for (int j = 0; j < 8; ++j) {
    qout[(size_t)(m0 + j + mhi) * H_DIM + (wave * 2 + 0) * 16 + nlo] = acc0[j];
    qout[(size_t)(m0 + j + mhi) * H_DIM + (wave * 2 + 1) * 16 + nlo] = acc1[j];
  }
}

// ---------------------------------------------------------------------------
// Kernel 2: 3-layer MLP over a 16-row object tile; emits k,v (bf16).
// X tile is brought into LDS by the Tensor Data Mover (wave 0 issues it;
// TDM pads each 128-half row by 8 halves to match the fragment pitch).
// ---------------------------------------------------------------------------
__global__ void mlp_kernel(const __bf16* __restrict__ xobj,
                           const __bf16* __restrict__ W1s, const float* __restrict__ b1,
                           const __bf16* __restrict__ W2s, const float* __restrict__ b2,
                           const __bf16* __restrict__ W3s, const float* __restrict__ b3,
                           __bf16* __restrict__ kout, __bf16* __restrict__ vout) {
  __shared__ __align__(16) __bf16 Xb[16][F_DIM + 8];        // 136 halves pitch
  __shared__ __align__(16) __bf16 Hb[2][16][H_DIM + 8];     // 264 halves pitch
  int tid = threadIdx.x, wave = tid >> 5, lane = tid & 31;
  int m0 = blockIdx.x * 16;
  int nlo = lane & 15;
  int mhi = (lane >> 4) << 3;

  if (wave == 0)
    tdm_load_x_tile(xobj + (size_t)m0 * F_DIM, (unsigned)(size_t)&Xb[0][0]);
  __builtin_amdgcn_s_wait_tensorcnt(0);
  __syncthreads();

  // ---- layer 1: K=128 (4 chunks), N=256 (wave -> 2 tiles) ----
  {
    v8f acc0, acc1;
    float bias0 = b1[(wave * 2 + 0) * 16 + nlo];
    float bias1 = b1[(wave * 2 + 1) * 16 + nlo];
#pragma unroll
    for (int j = 0; j < 8; ++j) { acc0[j] = bias0; acc1[j] = bias1; }
#pragma unroll
    for (int kt = 0; kt < 4; ++kt) {
      v16bf a = load_A_frag(&Xb[0][0], F_DIM + 8, lane, kt);
      acc0 = wmma_bf16(a, load_B_frag(W1s, 4, kt, wave * 2 + 0, lane), acc0);
      acc1 = wmma_bf16(a, load_B_frag(W1s, 4, kt, wave * 2 + 1, lane), acc1);
    }
#pragma unroll
    for (int j = 0; j < 8; ++j) {
      float x0 = acc0[j], x1 = acc1[j];
      Hb[0][j + mhi][(wave * 2 + 0) * 16 + nlo] = (__bf16)(x0 > 0.f ? x0 : 0.f);
      Hb[0][j + mhi][(wave * 2 + 1) * 16 + nlo] = (__bf16)(x1 > 0.f ? x1 : 0.f);
    }
  }
  __syncthreads();

  // ---- layer 2: K=256 (8 chunks), N=256 ----
  {
    v8f acc0, acc1;
    float bias0 = b2[(wave * 2 + 0) * 16 + nlo];
    float bias1 = b2[(wave * 2 + 1) * 16 + nlo];
#pragma unroll
    for (int j = 0; j < 8; ++j) { acc0[j] = bias0; acc1[j] = bias1; }
#pragma unroll
    for (int kt = 0; kt < 8; ++kt) {
      v16bf a = load_A_frag(&Hb[0][0][0], H_DIM + 8, lane, kt);
      acc0 = wmma_bf16(a, load_B_frag(W2s, 8, kt, wave * 2 + 0, lane), acc0);
      acc1 = wmma_bf16(a, load_B_frag(W2s, 8, kt, wave * 2 + 1, lane), acc1);
    }
#pragma unroll
    for (int j = 0; j < 8; ++j) {
      float x0 = acc0[j], x1 = acc1[j];
      Hb[1][j + mhi][(wave * 2 + 0) * 16 + nlo] = (__bf16)(x0 > 0.f ? x0 : 0.f);
      Hb[1][j + mhi][(wave * 2 + 1) * 16 + nlo] = (__bf16)(x1 > 0.f ? x1 : 0.f);
    }
  }
  __syncthreads();

  // ---- layer 3: K=256 (8 chunks), N=512 (wave -> 4 tiles) ----
  {
    v8f acc[4];
#pragma unroll
    for (int s = 0; s < 4; ++s) {
      float bias = b3[(wave * 4 + s) * 16 + nlo];
#pragma unroll
      for (int j = 0; j < 8; ++j) acc[s][j] = bias;
    }
#pragma unroll
    for (int kt = 0; kt < 8; ++kt) {
      v16bf a = load_A_frag(&Hb[1][0][0], H_DIM + 8, lane, kt);
#pragma unroll
      for (int s = 0; s < 4; ++s)
        acc[s] = wmma_bf16(a, load_B_frag(W3s, 8, kt, wave * 4 + s, lane), acc[s]);
    }
#pragma unroll
    for (int s = 0; s < 4; ++s) {
      int nt = wave * 4 + s;                     // 0..31
      int n  = nt * 16 + nlo;
      if (nt < 16) {                             // k half (N 0..255)
#pragma unroll
        for (int j = 0; j < 8; ++j)
          kout[(size_t)(m0 + j + mhi) * H_DIM + n] = (__bf16)acc[s][j];
      } else {                                   // v half (N 256..511)
        int nv = n - H_DIM;
#pragma unroll
        for (int j = 0; j < 8; ++j)
          vout[(size_t)(m0 + j + mhi) * H_DIM + nv] = (__bf16)acc[s][j];
      }
    }
  }
}

// ---------------------------------------------------------------------------
// Kernel 3: logits[t] = (k[t] . q[seg[t]]) / 16    (one wave per row)
// ---------------------------------------------------------------------------
__global__ void logits_kernel(const __bf16* __restrict__ kmat,
                              const float* __restrict__ qf,
                              const int* __restrict__ seg,
                              float* __restrict__ logits) {
  int wave = threadIdx.x >> 5, lane = threadIdx.x & 31;
  int t = blockIdx.x * 8 + wave;
  int s = seg[t];
  const __bf16* krow = kmat + (size_t)t * H_DIM;
  const float*  qrow = qf + (size_t)s * H_DIM;
  float acc = 0.f;
#pragma unroll
  for (int c = lane; c < H_DIM; c += 32)
    acc += (float)krow[c] * qrow[c];
#pragma unroll
  for (int off = 16; off > 0; off >>= 1)
    acc += __shfl_down(acc, off, 32);
  if (lane == 0) logits[t] = acc * 0.0625f;   // 1/sqrt(256)
}

// ---------------------------------------------------------------------------
// Kernel 4: per-segment softmax + weighted sum of v. One block per segment.
// ---------------------------------------------------------------------------
__global__ void segment_kernel(const float* __restrict__ logits,
                               const __bf16* __restrict__ vmat,
                               const int* __restrict__ seg,
                               float* __restrict__ emb_out,
                               float* __restrict__ w_out,
                               int T) {
  __shared__ int sh_start, sh_end;
  __shared__ float red[256];
  __shared__ float sh_m, sh_z;
  int s = blockIdx.x, tid = threadIdx.x;

  if (tid == 0) {
    int lo = 0, hi = T;
    while (lo < hi) { int mid = (lo + hi) >> 1; if (seg[mid] < s) lo = mid + 1; else hi = mid; }
    sh_start = lo;
    hi = T;
    while (lo < hi) { int mid = (lo + hi) >> 1; if (seg[mid] < s + 1) lo = mid + 1; else hi = mid; }
    sh_end = lo;
  }
  __syncthreads();
  int start = sh_start, end = sh_end;

  // segment max
  float m = -INFINITY;
  for (int t = start + tid; t < end; t += 256) m = fmaxf(m, logits[t]);
  red[tid] = m; __syncthreads();
  for (int off = 128; off > 0; off >>= 1) {
    if (tid < off) red[tid] = fmaxf(red[tid], red[tid + off]);
    __syncthreads();
  }
  if (tid == 0) sh_m = red[0];
  __syncthreads();
  m = sh_m;

  // segment sum of exp
  float z = 0.f;
  for (int t = start + tid; t < end; t += 256) z += __expf(logits[t] - m);
  red[tid] = z; __syncthreads();
  for (int off = 128; off > 0; off >>= 1) {
    if (tid < off) red[tid] += red[tid + off];
    __syncthreads();
  }
  if (tid == 0) sh_z = red[0];
  __syncthreads();
  float invz = 1.f / sh_z;

  // write normalized weights
  for (int t = start + tid; t < end; t += 256)
    w_out[t] = __expf(logits[t] - m) * invz;

  // weighted sum of v: thread owns column h = tid
  float acc = 0.f;
  for (int t = start; t < end; ++t) {
    float wt = __expf(logits[t] - m) * invz;
    acc += wt * (float)vmat[(size_t)t * H_DIM + tid];
  }
  emb_out[(size_t)s * H_DIM + tid] = acc;
}

// ---------------------------------------------------------------------------
// Host launcher
// ---------------------------------------------------------------------------
extern "C" void kernel_launch(void* const* d_in, const int* in_sizes, int n_in,
                              void* d_out, int out_size, void* d_ws, size_t ws_size,
                              hipStream_t stream) {
  const float* objects = (const float*)d_in[0];   // [T, F]
  const float* context = (const float*)d_in[1];   // [B, C]
  const int*   segids  = (const int*)  d_in[2];   // [T]
  const float* W1 = (const float*)d_in[3];        // [F, H]
  const float* b1 = (const float*)d_in[4];
  const float* W2 = (const float*)d_in[5];        // [H, H]
  const float* b2 = (const float*)d_in[6];
  const float* W3 = (const float*)d_in[7];        // [H, 2H]
  const float* b3 = (const float*)d_in[8];
  const float* Wq = (const float*)d_in[9];        // [C, H]
  const float* bq = (const float*)d_in[10];

  float* out = (float*)d_out;
  float* emb_out = out;                           // [B, H]
  float* w_out   = out + (size_t)B_SEG * H_DIM;   // [T]

  // workspace carve-out
  char* p = (char*)d_ws;
  float*  qf     = (float*)p;  p += (size_t)B_SEG * H_DIM * 4;
  __bf16* kbuf   = (__bf16*)p; p += (size_t)T_OBJ * H_DIM * 2;
  __bf16* vbuf   = (__bf16*)p; p += (size_t)T_OBJ * H_DIM * 2;
  float*  logits = (float*)p;  p += (size_t)T_OBJ * 4;
  __bf16* xbf    = (__bf16*)p; p += (size_t)T_OBJ * F_DIM * 2;
  __bf16* W1s    = (__bf16*)p; p += (size_t)F_DIM * H_DIM * 2;
  __bf16* W2s    = (__bf16*)p; p += (size_t)H_DIM * H_DIM * 2;
  __bf16* W3s    = (__bf16*)p; p += (size_t)H_DIM * 2 * H_DIM * 2;
  __bf16* Wqs    = (__bf16*)p; p += (size_t)C_DIM * H_DIM * 2;

  // A) objects fp32 -> bf16 (row-major, for TDM tile loads)
  {
    int n4 = (T_OBJ * F_DIM) / 4;                 // 8,388,608
    cvt_bf16_kernel<<<(n4 + 255) / 256, 256, 0, stream>>>((const float4*)objects,
                                                          xbf, n4);
  }

  // 0) weight swizzle+convert
  {
    int t1 = (F_DIM / 32) * (H_DIM / 16) * 32;        // 2048
    int t2 = (H_DIM / 32) * (H_DIM / 16) * 32;        // 4096
    int t3 = (H_DIM / 32) * (2 * H_DIM / 16) * 32;    // 8192
    int tq = (C_DIM / 32) * (H_DIM / 16) * 32;        // 8192
    swizzle_weights_kernel<<<(t1 + 255) / 256, 256, 0, stream>>>(W1, W1s, F_DIM, H_DIM);
    swizzle_weights_kernel<<<(t2 + 255) / 256, 256, 0, stream>>>(W2, W2s, H_DIM, H_DIM);
    swizzle_weights_kernel<<<(t3 + 255) / 256, 256, 0, stream>>>(W3, W3s, H_DIM, 2 * H_DIM);
    swizzle_weights_kernel<<<(tq + 255) / 256, 256, 0, stream>>>(Wq, Wqs, C_DIM, H_DIM);
  }

  // 1) q = context @ Wq + bq
  qgemm_kernel<<<B_SEG / 16, 256, 0, stream>>>(context, Wqs, bq, qf);

  // 2) MLP -> k, v (TDM-fed)
  mlp_kernel<<<T_OBJ / 16, 256, 0, stream>>>(xbf, W1s, b1, W2s, b2, W3s, b3,
                                             kbuf, vbuf);

  // 3) logits
  logits_kernel<<<T_OBJ / 8, 256, 0, stream>>>(kbuf, qf, segids, logits);

  // 4) per-segment softmax + weighted value sum
  segment_kernel<<<B_SEG, 256, 0, stream>>>(logits, vbuf, segids, emb_out, w_out, T_OBJ);
}
